// TDMHSA_9096740733548
// MI455X (gfx1250) — matrix operand order repaired and checked by reading, not compiled, not run on hardware
//
#include <hip/hip_runtime.h>
#include <hip/hip_bf16.h>

// ---------------------------------------------------------------------------
// Fused windowed MHSA for MI455X (gfx1250, wave32, WMMA bf16 16x16x32).
//
// v3: attention phases (QK^T and PV) now hoist all 8 B fragments and use 4
// interleaved accumulators -> no dependent back-to-back v_wmma, one dscnt
// wait per tile-quad instead of four, D->VALU hazard nops amortized.
// (v2: weight-stationary global GEMMs -> Wqkv/Wo read from L2 exactly once
// per block; two heads per pass for perfect 8-wave balance.)
// ---------------------------------------------------------------------------

typedef __bf16 v16bf __attribute__((ext_vector_type(16)));
typedef __bf16 v8bf  __attribute__((ext_vector_type(8)));
typedef float  v8f   __attribute__((ext_vector_type(8)));

// ---- geometry -------------------------------------------------------------
constexpr int C_     = 512;
constexpr int H_     = 8;
constexpr int DH     = 64;      // head dim
constexpr int NWIN   = 1024;    // B * 8 * 8
constexpr int NT     = 64;      // tokens per window
constexpr float SCALE = 0.125f; // 64^-0.5

// ---- LDS layout (bytes, all 32B aligned) ----------------------------------
constexpr int XS = 528;   // halves stride for 64x512 bf16 tiles (+16 pad)
constexpr int QS = 80;    // halves stride for 64x64  bf16 tiles (+16 pad)
constexpr int SS = 68;    // float stride for score rows
constexpr int OFF_X    = 0;                          // x tile      (bf16 64xXS)
constexpr int OFF_O    = OFF_X + 64 * XS * 2;        // attn out    (bf16 64xXS)
constexpr int OFF_Q    = OFF_O + 64 * XS * 2;        // q, 2 heads  (bf16 2x64xQS)
constexpr int OFF_K    = OFF_Q + 2 * 64 * QS * 2;    // k, 2 heads
constexpr int OFF_V    = OFF_K + 2 * 64 * QS * 2;    // vT, 2 heads [d][tok]
constexpr int OFF_P    = OFF_V + 2 * 64 * QS * 2;    // probs, 2 heads (bf16)
constexpr int OFF_S    = OFF_P + 2 * 64 * QS * 2;    // scores, 2 heads (f32)
constexpr int OFF_BIAS = OFF_S + 2 * 64 * SS * 4;    // rel-pos table (225x8 f32)
constexpr int SMEM_BYTES = OFF_BIAS + 225 * 8 * 4;   // = 259104 B (< 320KB/WGP)

// ---- WMMA fragment helpers (contiguous k order, same for A and B) ------------------
// A 16x32 bf16: lane m = lane&15, kbA = (lane&16)?16:0;
//   halves j -> K=kbA+j  => two contiguous 16B loads.
__device__ __forceinline__ v16bf loadA16(const __bf16* p) {
  v8bf lo = *(const v8bf*)p;
  v8bf hi = *(const v8bf*)(p + 8);
  return __builtin_shufflevector(lo, hi, 0,1,2,3,4,5,6,7,8,9,10,11,12,13,14,15);
}
// B 32x16 bf16: lane n = lane&15, kbB = (lane&16)?16:0; halves j -> K=kbB+j
//   => one contiguous 32B load when memory is [n][k].
__device__ __forceinline__ v8f wmma_bf16(v16bf a, v16bf b, v8f c) {
  return __builtin_amdgcn_wmma_f32_16x16x32_bf16(false, a, false, b, (short)0, c,
                                                 false, false);
}

// ---------------------------------------------------------------------------
// Prep kernel 1: fp32 weights -> bf16 in workspace.
// ---------------------------------------------------------------------------
__global__ __launch_bounds__(256) void wattn_prep_convert(
    const float* __restrict__ wqkv, const float* __restrict__ wo,
    __bf16* __restrict__ wqkv_bf, __bf16* __restrict__ wo_bf) {
  int i = blockIdx.x * 256 + threadIdx.x;
  if (i < 3 * C_ * C_) wqkv_bf[i] = (__bf16)wqkv[i];
  if (i < C_ * C_)     wo_bf[i]   = (__bf16)wo[i];
}

// ---------------------------------------------------------------------------
// Prep kernel 2: tadd[b][o] = bqkv[o] + bt[o] + dot(time_emb[b], Wt[o])
// ---------------------------------------------------------------------------
__global__ __launch_bounds__(256) void wattn_prep_tadd(
    const float* __restrict__ te, const float* __restrict__ wt,
    const float* __restrict__ bt, const float* __restrict__ bqkv,
    float* __restrict__ tadd) {
  int i = blockIdx.x * 256 + threadIdx.x;
  if (i >= 16 * 3 * C_) return;
  int b = i / (3 * C_), o = i % (3 * C_);
  const float* t = te + b * C_;
  const float* w = wt + o * C_;
  float s = bqkv[o] + bt[o];
  #pragma unroll 8
  for (int k = 0; k < C_; ++k) s += t[k] * w[k];
  tadd[i] = s;
}

// ---------------------------------------------------------------------------
// Main fused kernel: one block per window, 256 threads (8 waves).
// ---------------------------------------------------------------------------
__global__ __launch_bounds__(256) void wattn_fused(
    const float* __restrict__ x, const float* __restrict__ bias_table,
    const __bf16* __restrict__ wqkv, const __bf16* __restrict__ wo,
    const float* __restrict__ tadd, const float* __restrict__ bo,
    float* __restrict__ out) {
  extern __shared__ char smem[];
  __bf16* s_x    = (__bf16*)(smem + OFF_X);
  __bf16* s_o    = (__bf16*)(smem + OFF_O);
  __bf16* s_q    = (__bf16*)(smem + OFF_Q);   // [2][64][QS]
  __bf16* s_k    = (__bf16*)(smem + OFF_K);   // [2][64][QS]
  __bf16* s_vT   = (__bf16*)(smem + OFF_V);   // [2][64(d)][QS(tok)]
  __bf16* s_p    = (__bf16*)(smem + OFF_P);   // [2][64][QS]
  float*  s_s    = (float*) (smem + OFF_S);   // [2][64][SS]
  float*  s_bias = (float*) (smem + OFF_BIAS);

  const int tid  = threadIdx.x;
  const int lane = tid & 31;
  const int wv   = tid >> 5;
  const int ln   = lane & 15;
  const int hi8  = (lane & 16) ? 8 : 0;   // C-frag row offset
  const int kbA  = (lane & 16) ? 16 : 0;  // A-frag K base
  const int kbB  = (lane & 16) ? 16 : 0;  // B-frag K base

  const int win  = blockIdx.x;
  const int bimg = win >> 6;
  const int widx = win & 63;
  const int nh   = widx >> 3, nw = widx & 7;

  // ---- Phase 0: stage rel-pos table + window of x (fp32->bf16) into LDS ---
  for (int i = tid; i < 225 * 8; i += 256) s_bias[i] = bias_table[i];
  {
    const int m  = tid >> 2;           // token row 0..63
    const int q4 = tid & 3;            // 128-col quarter
    const int wy = m >> 3, wx = m & 7;
    const int gtok = ((nh * 8 + wy) << 6) + (nw * 8 + wx);
    const float* xr = x + ((size_t)bimg * 4096 + gtok) * C_ + q4 * 128;
    __bf16* dst = s_x + m * XS + q4 * 128;
    #pragma unroll
    for (int j = 0; j < 128; j += 8) {
      float4 f0 = *(const float4*)(xr + j);
      float4 f1 = *(const float4*)(xr + j + 4);
      v8bf v;
      v[0] = (__bf16)f0.x; v[1] = (__bf16)f0.y; v[2] = (__bf16)f0.z; v[3] = (__bf16)f0.w;
      v[4] = (__bf16)f1.x; v[5] = (__bf16)f1.y; v[6] = (__bf16)f1.z; v[7] = (__bf16)f1.w;
      *(v8bf*)(dst + j) = v;
    }
  }
  __syncthreads();

  const __bf16* abase_x = s_x + ln * XS + kbA;   // + mi*16*XS + k0

  // ---- Head-pair pipeline (4 passes of 2 heads) ---------------------------
  for (int hp = 0; hp < 4; ++hp) {
    // Phase 1: QKV for 2 heads: (64x512)@(512x384). 24 N-column jobs, 3/wave.
    // Weight-stationary: one B load per K-step feeds 4 M accumulators.
    #pragma unroll
    for (int j = 0; j < 3; ++j) {
      const int nj  = wv + j * 8;        // 0..23
      const int hh  = nj / 12;           // head within pair
      const int rem = nj % 12;
      const int s   = rem >> 2, dt = rem & 3;   // q/k/v select, d-tile
      const int h   = hp * 2 + hh;
      const __bf16* bbase = wqkv + (size_t)(s * C_ + h * DH + dt * 16 + ln) * C_ + kbB;
      v8f acc[4] = {{}, {}, {}, {}};
      #pragma unroll 4
      for (int k0 = 0; k0 < C_; k0 += 32) {
        if ((k0 & 63) == 0) __builtin_prefetch(bbase + k0 + 128, 0, 0);
        const v16bf b = *(const v16bf*)(bbase + k0);
        #pragma unroll
        for (int mi = 0; mi < 4; ++mi)
          acc[mi] = wmma_bf16(loadA16(abase_x + mi * 16 * XS + k0), b, acc[mi]);
      }
      const int d  = dt * 16 + ln;
      const float ta = tadd[bimg * (3 * C_) + s * C_ + h * DH + d];
      __bf16* qk = (s == 0 ? s_q : s_k) + hh * 64 * QS;
      #pragma unroll
      for (int mi = 0; mi < 4; ++mi) {
        #pragma unroll
        for (int r = 0; r < 8; ++r) {
          const int m = mi * 16 + r + hi8;
          const __bf16 v = (__bf16)(acc[mi][r] + ta);
          if (s < 2) qk[m * QS + d] = v;
          else       s_vT[hh * 64 * QS + d * QS + m] = v;  // V transposed
        }
      }
    }
    __syncthreads();

    // Phase 2a: scores = q @ k^T * SCALE + bias.  32 tiles, 4/wave.
    // Fixed (hh, mi) per wave; hoist A frags + all 8 B frags, interleave
    // 4 accumulators -> independent WMMA stream, single dscnt wait.
    {
      const int hh = wv >> 2, mi = wv & 3;
      const int h  = hp * 2 + hh;
      const __bf16* aq = s_q + hh * 64 * QS + (mi * 16 + ln) * QS + kbA;
      const v16bf a0 = loadA16(aq);
      const v16bf a1 = loadA16(aq + 32);
      const __bf16* kb = s_k + hh * 64 * QS + ln * QS + kbB;
      v16bf b0[4], b1[4];
      #pragma unroll
      for (int njj = 0; njj < 4; ++njj) {
        b0[njj] = *(const v16bf*)(kb + njj * 16 * QS);
        b1[njj] = *(const v16bf*)(kb + njj * 16 * QS + 32);
      }
      v8f acc[4];
      const v8f z = {};
      #pragma unroll
      for (int njj = 0; njj < 4; ++njj) acc[njj] = wmma_bf16(a0, b0[njj], z);
      #pragma unroll
      for (int njj = 0; njj < 4; ++njj) acc[njj] = wmma_bf16(a1, b1[njj], acc[njj]);
      #pragma unroll
      for (int njj = 0; njj < 4; ++njj) {
        const int jc = njj * 16 + ln;
        const int yj = jc >> 3, xj = jc & 7;
        #pragma unroll
        for (int r = 0; r < 8; ++r) {
          const int ic = mi * 16 + r + hi8;
          const int yi = ic >> 3, xi = ic & 7;
          const int idx = (yi - yj + 7) * 15 + (xi - xj + 7);
          s_s[hh * 64 * SS + ic * SS + jc] = acc[njj][r] * SCALE + s_bias[idx * H_ + h];
        }
      }
    }
    __syncthreads();

    // Phase 2b: row softmax, 128 rows (2 heads x 64), one thread per row.
    if (tid < 2 * NT) {
      const int hh = tid >> 6, r = tid & 63;
      float* row = s_s + hh * 64 * SS + r * SS;
      float mx = -1e30f;
      #pragma unroll 8
      for (int jj = 0; jj < NT; ++jj) mx = fmaxf(mx, row[jj]);
      float sum = 0.f;
      #pragma unroll 8
      for (int jj = 0; jj < NT; ++jj) { float e = __expf(row[jj] - mx); row[jj] = e; sum += e; }
      const float inv = 1.f / sum;
      __bf16* pr = s_p + hh * 64 * QS + r * QS;
      #pragma unroll 8
      for (int jj = 0; jj < NT; ++jj) pr[jj] = (__bf16)(row[jj] * inv);
    }
    __syncthreads();

    // Phase 2c: O_h = P @ V.  Same interleaved-quad structure as 2a.
    {
      const int hh = wv >> 2, mi = wv & 3;
      const int h  = hp * 2 + hh;
      const __bf16* ap = s_p + hh * 64 * QS + (mi * 16 + ln) * QS + kbA;
      const v16bf a0 = loadA16(ap);
      const v16bf a1 = loadA16(ap + 32);
      const __bf16* vb = s_vT + hh * 64 * QS + ln * QS + kbB;
      v16bf b0[4], b1[4];
      #pragma unroll
      for (int njj = 0; njj < 4; ++njj) {
        b0[njj] = *(const v16bf*)(vb + njj * 16 * QS);
        b1[njj] = *(const v16bf*)(vb + njj * 16 * QS + 32);
      }
      v8f acc[4];
      const v8f z = {};
      #pragma unroll
      for (int njj = 0; njj < 4; ++njj) acc[njj] = wmma_bf16(a0, b0[njj], z);
      #pragma unroll
      for (int njj = 0; njj < 4; ++njj) acc[njj] = wmma_bf16(a1, b1[njj], acc[njj]);
      #pragma unroll
      for (int njj = 0; njj < 4; ++njj) {
        #pragma unroll
        for (int r = 0; r < 8; ++r) {
          const int m = mi * 16 + r + hi8;
          s_o[m * XS + h * DH + njj * 16 + ln] = (__bf16)acc[njj][r];
        }
      }
    }
    __syncthreads();
  }

  // ---- Phase 3: out = O @ Wo^T + bo.  32 N-columns, 4/wave, B loaded once.
  const __bf16* abase_o = s_o + ln * XS + kbA;
  #pragma unroll
  for (int j = 0; j < 4; ++j) {
    const int nj = wv + j * 8;   // 0..31
    const __bf16* bbase = wo + (size_t)(nj * 16 + ln) * C_ + kbB;
    v8f acc[4] = {{}, {}, {}, {}};
    #pragma unroll 4
    for (int k0 = 0; k0 < C_; k0 += 32) {
      if ((k0 & 63) == 0) __builtin_prefetch(bbase + k0 + 128, 0, 0);
      const v16bf b = *(const v16bf*)(bbase + k0);
      #pragma unroll
      for (int mi = 0; mi < 4; ++mi)
        acc[mi] = wmma_bf16(loadA16(abase_o + mi * 16 * XS + k0), b, acc[mi]);
    }
    const int col = nj * 16 + ln;
    const float bb = bo[col];
    #pragma unroll
    for (int mi = 0; mi < 4; ++mi) {
      #pragma unroll
      for (int r = 0; r < 8; ++r) {
        const int m = mi * 16 + r + hi8;
        out[((size_t)win * NT + m) * C_ + col] = acc[mi][r] + bb;
      }
    }
  }
}

// ---------------------------------------------------------------------------
extern "C" void kernel_launch(void* const* d_in, const int* in_sizes, int n_in,
                              void* d_out, int out_size, void* d_ws, size_t ws_size,
                              hipStream_t stream) {
  const float* x          = (const float*)d_in[0];
  const float* time_emb   = (const float*)d_in[1];
  const float* Wqkv       = (const float*)d_in[2];
  const float* bqkv       = (const float*)d_in[3];
  const float* Wt         = (const float*)d_in[4];
  const float* bt         = (const float*)d_in[5];
  const float* Wo         = (const float*)d_in[6];
  const float* bo         = (const float*)d_in[7];
  const float* bias_table = (const float*)d_in[8];
  float* out = (float*)d_out;

  // Workspace carve-up (~2.2 MB total): bf16 weights + fp32 time addend.
  __bf16* wqkv_bf = (__bf16*)d_ws;                         // 1536*512*2 B
  __bf16* wo_bf   = wqkv_bf + 3 * C_ * C_;                 //  512*512*2 B
  float*  tadd    = (float*)(wo_bf + C_ * C_);             // 16*1536*4  B

  wattn_prep_convert<<<(3 * C_ * C_ + 255) / 256, 256, 0, stream>>>(Wqkv, Wo, wqkv_bf, wo_bf);
  wattn_prep_tadd<<<(16 * 3 * C_ + 255) / 256, 256, 0, stream>>>(time_emb, Wt, bt, bqkv, tadd);
  wattn_fused<<<NWIN, 256, SMEM_BYTES, stream>>>(x, bias_table, wqkv_bf, wo_bf, tadd, bo, out);
}
